// HarmonicConv_73332271611976
// MI455X (gfx1250) — matrix operand order repaired and checked by reading, not compiled, not run on hardware
//
#include <hip/hip_runtime.h>
#include <hip/hip_bf16.h>
#include <stdint.h>

// Problem constants (match reference)
#define B_    8
#define CIN_  28
#define COUT_ 256
#define H_    256
#define W_    256
#define DG_   7
#define KH_   7
#define CPG_  4
#define K1_   196   // DG*CPG*KH
#define K1P_  224   // padded to 7 chunks of 32

// LDS K-pitch in halves: 132 dwords == 4 (mod 64 banks) -> conflict-free b128 reads
#define KPITCH_ 264

typedef __attribute__((ext_vector_type(16))) __bf16 v16bf;
typedef __attribute__((ext_vector_type(8)))  float  v8f;

union FragAB { v16bf v; unsigned short s[16]; uint4 q[2]; };
union FragC  { v8f v; float a[8]; };

__device__ __forceinline__ unsigned short f2bf(float f) {
  // round-to-nearest-even f32 -> bf16
  unsigned u = __float_as_uint(f);
  u += 0x7FFFu + ((u >> 16) & 1u);
  return (unsigned short)(u >> 16);
}

// A-fragment K index in this file's packing (contiguous order, same as the B loads):
// lanes 0-15 hold K {0..15}; lanes 16-31 hold K {16..31}.
__device__ __forceinline__ int a_krel(int lane, int j) {
  return ((lane >= 16) ? 16 : 0) + j;
}

// ---------------------------------------------------------------------------
// Kernel 0: pack weights into per-lane WMMA A-fragment order (bf16)
// Apack1: [kk(7)][otile(16)][lane(32)][j(16)]       from weight (COUT,196)
// Apack2: [i(7)][cc(8)][otile(16)][lane(32)][j(16)] from w_t (COUT,COUT,1,7)
// ---------------------------------------------------------------------------
__global__ __launch_bounds__(256) void pack_weights(
    const float* __restrict__ weight, const float* __restrict__ w_t,
    unsigned short* __restrict__ Apack1, unsigned short* __restrict__ Apack2) {
  int tid = blockIdx.x * 256 + threadIdx.x;
  if (tid < 7 * 16 * 32 * 16) {  // 57344
    int j = tid & 15, lane = (tid >> 4) & 31, ot = (tid >> 9) & 15, kk = tid >> 13;
    int o = ot * 16 + (lane & 15);
    int k = kk * 32 + a_krel(lane, j);
    float v = (k < K1_) ? weight[o * K1_ + k] : 0.0f;
    Apack1[tid] = f2bf(v);
  }
  if (tid < 7 * 8 * 16 * 32 * 16) {  // 458752
    int j = tid & 15, lane = (tid >> 4) & 31, ot = (tid >> 9) & 15;
    int cc = (tid >> 13) & 7, i = tid >> 16;
    int o = ot * 16 + (lane & 15);
    int c = cc * 32 + a_krel(lane, j);
    Apack2[tid] = f2bf(w_t[(o * COUT_ + c) * KH_ + i]);
  }
}

// ---------------------------------------------------------------------------
// Kernel 1: harmonic vertical sampling + grouped conv as bf16 WMMA GEMM.
// Block = one (b,h), 32 w's, all 256 couts. out1 bf16 layout [b][h][c][w].
// Sampled slab staged COLUMN-major S2[n][k] so B fragments are contiguous.
// ---------------------------------------------------------------------------
__global__ __launch_bounds__(256) void stage1_sample_conv(
    const float* __restrict__ x, const unsigned short* __restrict__ Apack1,
    const float* __restrict__ bias, unsigned short* __restrict__ out1) {
  __shared__ __align__(16) unsigned short S2[32][KPITCH_];  // [n][k], 16.9 KB
  __shared__ int   y0s[49], y1s[49];
  __shared__ float w0s[49], w1s[49];

  const int w0g = blockIdx.x * 32;
  const int h   = blockIdx.y;
  const int b   = blockIdx.z;
  const int tid = threadIdx.x;

  if (tid < 49) {                            // per-(g,i) sampling params
    int g = tid / 7, i = tid % 7;
    // yp = h - PADH + i + off simplifies to (h+1)(i+1)/(g+1) - 4
    float yp = (float)(h + 1) * (float)(i + 1) / (float)(g + 1) - 4.0f;
    float y0f = floorf(yp);
    float fy  = yp - y0f;
    int   y0  = (int)y0f;
    float v0  = (y0 >= 0 && y0 <= H_ - 1) ? 1.0f : 0.0f;
    float v1  = (y0 + 1 >= 0 && y0 + 1 <= H_ - 1) ? 1.0f : 0.0f;
    y0s[tid] = min(max(y0, 0), H_ - 1);
    y1s[tid] = min(max(y0 + 1, 0), H_ - 1);
    w0s[tid] = (1.0f - fy) * v0;
    w1s[tid] = fy * v1;
  }
  __syncthreads();

  // Build S2[n][k]: bilinear blend of two x rows (coalesced in w), f32 -> bf16
  const int col = tid & 31;
  for (int r = (tid >> 5); r < K1P_; r += 8) {
    unsigned short val = 0;
    if (r < K1_) {
      int g = r / 28, rem = r % 28, c = rem / 7, i = rem % 7;
      int gi = g * 7 + i;
      const float* xb = x + (((size_t)b * CIN_ + (g * CPG_ + c)) * H_) * W_;
      float v = w0s[gi] * xb[(size_t)y0s[gi] * W_ + w0g + col]
              + w1s[gi] * xb[(size_t)y1s[gi] * W_ + w0g + col];
      val = f2bf(v);
    }
    S2[col][r] = val;
  }
  __syncthreads();

  const int wave = tid >> 5, lane = tid & 31;
  FragC acc[2][2];
#pragma unroll
  for (int a = 0; a < 2; ++a)
#pragma unroll
    for (int n = 0; n < 2; ++n)
#pragma unroll
      for (int r = 0; r < 8; ++r) acc[a][n].a[r] = 0.0f;

  for (int kk = 0; kk < 7; ++kk) {
    // B fragment: lane<16 -> N=lane, K kb..kb+15; lane>=16 -> N=lane-16, K+16
    const int kb = kk * 32 + ((lane >> 4) << 4);
    FragAB fa[2];
#pragma unroll
    for (int ot2 = 0; ot2 < 2; ++ot2) {
      const int ot = wave * 2 + ot2;
      const uint4* ap = (const uint4*)(Apack1 + ((((size_t)kk * 16 + ot) * 32 + lane) << 4));
      fa[ot2].q[0] = ap[0]; fa[ot2].q[1] = ap[1];
    }
    FragAB fb[2];
#pragma unroll
    for (int nt = 0; nt < 2; ++nt) {
      const uint4* bp = (const uint4*)&S2[(lane & 15) + nt * 16][kb];  // 32B contiguous
      fb[nt].q[0] = bp[0]; fb[nt].q[1] = bp[1];
    }
#pragma unroll
    for (int ot2 = 0; ot2 < 2; ++ot2)
#pragma unroll
      for (int nt = 0; nt < 2; ++nt)
        acc[ot2][nt].v = __builtin_amdgcn_wmma_f32_16x16x32_bf16(
            false, fa[ot2].v, false, fb[nt].v, (short)0, acc[ot2][nt].v, false, false);
  }

  // Epilogue: +bias, f32 -> bf16, store out1[b][h][o][w]
  unsigned short* dst = out1 + ((size_t)(b * H_ + h)) * COUT_ * W_;
  const int m0 = (lane >= 16) ? 8 : 0;
#pragma unroll
  for (int ot2 = 0; ot2 < 2; ++ot2) {
    const int o0 = (wave * 2 + ot2) * 16 + m0;
#pragma unroll
    for (int nt = 0; nt < 2; ++nt) {
      const int w = w0g + (lane & 15) + nt * 16;
#pragma unroll
      for (int r = 0; r < 8; ++r) {
        const int o = o0 + r;
        dst[(size_t)o * W_ + w] = f2bf(acc[ot2][nt].a[r] + bias[o]);
      }
    }
  }
}

// ---------------------------------------------------------------------------
// Kernel 2: 1x7 channel conv as bf16 WMMA GEMM (dominant stage, ~0.48 TFLOP).
// Block = one (b,h), 64 w's (+/-3 halo), all 256 couts.
// Activation slab staged COLUMN-major X2[col][c]; tap i shifts the row only.
// t (f32) layout [b][c][h][w] (matches reference flatten order).
// ---------------------------------------------------------------------------
__global__ __launch_bounds__(256) void stage2_hconv(
    const unsigned short* __restrict__ out1, const unsigned short* __restrict__ Apack2,
    const float* __restrict__ b_t, float* __restrict__ t) {
  __shared__ __align__(16) unsigned short X2[70][KPITCH_];  // [w-col][c], 37 KB

  const int w0g = blockIdx.x * 64;
  const int h   = blockIdx.y;
  const int b   = blockIdx.z;
  const int tid = threadIdx.x;

  // Stage: coalesced global reads (col fastest), transposed LDS writes
  const unsigned short* src = out1 + ((size_t)(b * H_ + h)) * COUT_ * W_;
  for (int idx = tid; idx < COUT_ * 70; idx += 256) {
    int c = idx / 70, col = idx % 70;
    int w = w0g - 3 + col;
    X2[col][c] = ((unsigned)w < (unsigned)W_) ? src[(size_t)c * W_ + w] : (unsigned short)0;
  }
  __syncthreads();

  const int wave = tid >> 5, lane = tid & 31;
  FragC acc[2][4];
#pragma unroll
  for (int a = 0; a < 2; ++a)
#pragma unroll
    for (int n = 0; n < 4; ++n)
#pragma unroll
      for (int r = 0; r < 8; ++r) acc[a][n].a[r] = 0.0f;

  for (int cc = 0; cc < 8; ++cc) {
    const int kb = cc * 32 + ((lane >> 4) << 4);
    // hint next chunk's A fragments toward the caches
    __builtin_prefetch(Apack2 + ((((size_t)((cc + 1) & 7)) * 16 + wave * 2) * 32 + lane) * 16, 0, 1);
    for (int i = 0; i < 7; ++i) {
      FragAB fa[2];
#pragma unroll
      for (int ot2 = 0; ot2 < 2; ++ot2) {
        const int ot = wave * 2 + ot2;
        const uint4* ap = (const uint4*)(Apack2 +
            (((((size_t)i * 8 + cc) * 16 + ot) * 32 + lane) << 4));
        fa[ot2].q[0] = ap[0]; fa[ot2].q[1] = ap[1];
      }
      FragAB fb[4];
#pragma unroll
      for (int nt = 0; nt < 4; ++nt) {
        const int col = (lane & 15) + nt * 16 + i;   // w = w0g + n + i - 3
        const uint4* bp = (const uint4*)&X2[col][kb];  // 32B contiguous
        fb[nt].q[0] = bp[0]; fb[nt].q[1] = bp[1];
      }
#pragma unroll
      for (int ot2 = 0; ot2 < 2; ++ot2)
#pragma unroll
        for (int nt = 0; nt < 4; ++nt)
          acc[ot2][nt].v = __builtin_amdgcn_wmma_f32_16x16x32_bf16(
              false, fa[ot2].v, false, fb[nt].v, (short)0, acc[ot2][nt].v, false, false);
    }
  }

  const int m0 = (lane >= 16) ? 8 : 0;
#pragma unroll
  for (int ot2 = 0; ot2 < 2; ++ot2) {
    const int o0 = (wave * 2 + ot2) * 16 + m0;
#pragma unroll
    for (int nt = 0; nt < 4; ++nt) {
      const int w = w0g + (lane & 15) + nt * 16;
#pragma unroll
      for (int r = 0; r < 8; ++r) {
        const int o = o0 + r;
        t[(((size_t)b * COUT_ + o) * H_ + h) * W_ + w] = acc[ot2][nt].a[r] + b_t[o];
      }
    }
  }
}

// ---------------------------------------------------------------------------
// Kernel 3: group mix across DG floor-chunks of the flattened (c,h,w) axis.
// Each thread reads the 7 group values once and writes all 7 mixed outputs.
// ---------------------------------------------------------------------------
__global__ __launch_bounds__(256) void stage3_mix(
    const float* __restrict__ t, const float* __restrict__ w_m,
    const float* __restrict__ b_m, float* __restrict__ out) {
  const size_t CHW = (size_t)COUT_ * H_ * W_;   // 16777216
  const size_t Lg  = CHW / DG_;                  // 2396745 (floor; 1-elem tail)
  const size_t r   = (size_t)blockIdx.x * 256 + threadIdx.x;
  const int    b   = blockIdx.y;
  const float* tb  = t + (size_t)b * CHW;
  float*       ob  = out + (size_t)b * CHW;

  if (r < Lg) {
    float in[DG_];
#pragma unroll
    for (int g = 0; g < DG_; ++g) in[g] = tb[(size_t)g * Lg + r];
#pragma unroll
    for (int f = 0; f < DG_; ++f) {
      float s = b_m[f];
#pragma unroll
      for (int g = 0; g < DG_; ++g) s += w_m[f * DG_ + g] * in[g];
      ob[(size_t)f * Lg + r] = s;
    }
  }
  const size_t tail0 = (size_t)DG_ * Lg;
  if (r < CHW - tail0) ob[tail0 + r] = tb[tail0 + r];  // pass tail through
}

// ---------------------------------------------------------------------------
extern "C" void kernel_launch(void* const* d_in, const int* in_sizes, int n_in,
                              void* d_out, int out_size, void* d_ws, size_t ws_size,
                              hipStream_t stream) {
  (void)in_sizes; (void)n_in; (void)out_size; (void)ws_size;
  const float* x      = (const float*)d_in[0];
  const float* weight = (const float*)d_in[1];
  const float* bias   = (const float*)d_in[2];
  const float* w_t    = (const float*)d_in[3];
  const float* b_t    = (const float*)d_in[4];
  const float* w_m    = (const float*)d_in[5];
  const float* b_m    = (const float*)d_in[6];
  float* out = (float*)d_out;

  // Workspace carve-up (all 256B-aligned):
  char* ws = (char*)d_ws;
  unsigned short* Apack1 = (unsigned short*)(ws);                     // 114,688 B
  unsigned short* Apack2 = (unsigned short*)(ws + 114688);            // 917,504 B
  unsigned short* out1   = (unsigned short*)(ws + 1032192);           // 268 MB bf16
  float*          t      = (float*)(ws + 1032192 + 268435456ull);     // 536 MB f32

  pack_weights<<<1792, 256, 0, stream>>>(weight, w_t, Apack1, Apack2);

  dim3 g1(W_ / 32, H_, B_);
  stage1_sample_conv<<<g1, 256, 0, stream>>>(x, Apack1, bias, out1);

  dim3 g2(W_ / 64, H_, B_);
  stage2_hconv<<<g2, 256, 0, stream>>>(out1, Apack2, b_t, t);

  const size_t Lg = ((size_t)COUT_ * H_ * W_) / DG_;
  dim3 g3((unsigned)((Lg + 255) / 256), B_);
  stage3_mix<<<g3, 256, 0, stream>>>(t, w_m, b_m, out);
}
